// GCN_17695265259557
// MI455X (gfx1250) — matrix operand order repaired and checked by reading, not compiled, not run on hardware
//
#include <hip/hip_runtime.h>

// ---------------- problem constants (match reference) ----------------
#define NN     100000      // nodes
#define EE     1600000     // edges
#define FF     128         // input features
#define DD     32          // hidden dim
#define GG     1000        // graphs
#define CC     10          // classes
#define NT     (NN / 16)   // 6250 row tiles of 16 (exact)
#define BN_EPS 1e-5f

typedef __attribute__((ext_vector_type(16))) __bf16 v16bf;
typedef __attribute__((ext_vector_type(8)))  float  v8f;

#define WMMA_BF16(a, b, c) \
  __builtin_amdgcn_wmma_f32_16x16x32_bf16(false, (a), false, (b), (short)0, (c), false, false)

// ---------------------------------------------------------------------
// Weight prep: fp32 [B][K][Nout] -> bf16 transposed [B][Nout][K]
// ---------------------------------------------------------------------
__global__ void k_transpose_bf16(const float* __restrict__ src, __bf16* __restrict__ dst,
                                 int B, int K, int Nout) {
  int idx = blockIdx.x * blockDim.x + threadIdx.x;
  int total = B * K * Nout;
  if (idx >= total) return;
  int b = idx / (K * Nout);
  int r = idx % (K * Nout);
  int n = r / K;
  int k = r % K;
  dst[(size_t)(b * Nout + n) * K + k] = (__bf16)src[(size_t)(b * K + k) * Nout + n];
}

__global__ void k_zero(float* __restrict__ p, int n) {
  int i = blockIdx.x * blockDim.x + threadIdx.x;
  if (i < n) p[i] = 0.0f;
}

// Identity BN-affine (for the first layer's input, which has no preceding BN).
__global__ void k_init_affine(float* __restrict__ scale, float* __restrict__ shift) {
  int c = threadIdx.x;
  if (c < DD) { scale[c] = 1.0f; shift[c] = 0.0f; }
}

// ---------------------------------------------------------------------
// u = x[N,128] @ W1a  (bf16 WMMA, K=128 = 4 k-steps, 2 col tiles)
// wt = W1a^T as bf16 [32][128]
// ---------------------------------------------------------------------
__global__ void k_gemm_xw(const float* __restrict__ x, const __bf16* __restrict__ wt,
                          float* __restrict__ out) {
  const int lane = threadIdx.x & 31;
  const int l15  = lane & 15;      // A: row-in-tile, B/D: col-in-tile
  const int hh   = lane >> 4;      // K-half selector

  // Preload B fragments: 4 k-steps x 2 col tiles.
  v16bf bf[4][2];
#pragma unroll
  for (int ct = 0; ct < 2; ++ct)
#pragma unroll
    for (int ks = 0; ks < 4; ++ks) {
      const __bf16* r = wt + (size_t)(ct * 16 + l15) * FF + ks * 32 + 8 * hh;
#pragma unroll
      for (int j = 0; j < 16; ++j) bf[ks][ct][j] = r[(j & 7) + 16 * (j >> 3)];
    }

  const int wid = blockIdx.x * (blockDim.x >> 5) + (threadIdx.x >> 5);
  const int nw  = gridDim.x * (blockDim.x >> 5);

  for (int tile = wid; tile < NT; tile += nw) {
    const float* xp = x + (size_t)(tile * 16 + l15) * FF;
    v8f acc0 = {}, acc1 = {};
#pragma unroll
    for (int ks = 0; ks < 4; ++ks) {
      const int base = ks * 32;
      v16bf a;
#pragma unroll
      for (int j = 0; j < 8; ++j) {
        a[j]     = (__bf16)xp[base + 8 * hh + j];
        a[8 + j] = (__bf16)xp[base + 16 + 8 * hh + j];
      }
      acc0 = WMMA_BF16(a, bf[ks][0], acc0);
      acc1 = WMMA_BF16(a, bf[ks][1], acc1);
    }
    float* op = out + (size_t)tile * 16 * DD;
#pragma unroll
    for (int i = 0; i < 8; ++i) {
      op[(i + 8 * hh) * DD + l15]      = acc0[i];
      op[(i + 8 * hh) * DD + 16 + l15] = acc1[i];
    }
  }
}

// ---------------------------------------------------------------------
// Edge scatter with fused BN affine on the gathered source:
//   agg[dst] += bn(h[src]) = h[src]*scale + shift    (wave per edge)
// ---------------------------------------------------------------------
__global__ void k_scatter(const int* __restrict__ src, const int* __restrict__ dst,
                          const float* __restrict__ h, float* __restrict__ agg,
                          const float* __restrict__ scale, const float* __restrict__ shift) {
  const int lane = threadIdx.x & 31;
  const float sc = scale[lane];
  const float sh = shift[lane];
  const int w  = (blockIdx.x * blockDim.x + threadIdx.x) >> 5;
  const int nw = (gridDim.x * blockDim.x) >> 5;
  for (int e = w; e < EE; e += nw) {
    const int s = src[e];
    const int d = dst[e];
    const float v = fmaf(h[(size_t)s * DD + lane], sc, sh);
    atomicAdd(&agg[(size_t)d * DD + lane], v);
  }
}

// ---------------------------------------------------------------------
// Fused GIN MLP + BN stats, with BN affine applied to hin on load.
//   hb = hin*scale + shift   (affine from the PREVIOUS layer's BN)
//   DO_GEMM1=true : t = hb+agg; r1 = relu(t@Wa + ba); out = relu(r1@Wb + bb)
//   DO_GEMM1=false: r1 = relu(hb+agg+ba);             out = relu(r1@Wb + bb)
//                   (layer 1 after the linearity trick: hin = x@W1a, affine=identity)
// Accumulates per-column sum / sumsq of `out` into stat_sum / stat_sq.
// ---------------------------------------------------------------------
#define LDSS 40   // padded LDS row stride (bf16 elems)

template <bool DO_GEMM1>
__global__ void k_gin_mlp(const float* __restrict__ hin, const float* __restrict__ agg,
                          const float* __restrict__ scale, const float* __restrict__ shift,
                          const __bf16* __restrict__ wa_t, const float* __restrict__ ba,
                          const __bf16* __restrict__ wb_t, const float* __restrict__ bb,
                          float* __restrict__ out,
                          float* __restrict__ stat_sum, float* __restrict__ stat_sq) {
  __shared__ __bf16 lds_r1[4][16 * LDSS];
  __shared__ float s_sum[DD], s_sq[DD];

  const int lane  = threadIdx.x & 31;
  const int wslot = threadIdx.x >> 5;
  const int l15   = lane & 15;
  const int hh    = lane >> 4;

  // Preload second-GEMM weight fragments (and first-GEMM if used).
  v16bf wbF[2];
#pragma unroll
  for (int ct = 0; ct < 2; ++ct) {
    const __bf16* r = wb_t + (size_t)(ct * 16 + l15) * DD + 8 * hh;
#pragma unroll
    for (int j = 0; j < 16; ++j) wbF[ct][j] = r[(j & 7) + 16 * (j >> 3)];
  }
  v16bf waF[2];
  if constexpr (DO_GEMM1) {
#pragma unroll
    for (int ct = 0; ct < 2; ++ct) {
      const __bf16* r = wa_t + (size_t)(ct * 16 + l15) * DD + 8 * hh;
#pragma unroll
      for (int j = 0; j < 16; ++j) waF[ct][j] = r[(j & 7) + 16 * (j >> 3)];
    }
  }
  const float ba_c0 = ba[l15], ba_c1 = ba[16 + l15];   // first-linear bias by out-col
  const float bb_c0 = bb[l15], bb_c1 = bb[16 + l15];

  // BN affine + first-stage bias at this lane's A-fragment K positions.
  float scK0[8], scK1[8], shK0[8], shK1[8], baK0[8], baK1[8];
#pragma unroll
  for (int j = 0; j < 8; ++j) {
    const int k0 = 8 * hh + j;
    const int k1 = 16 + 8 * hh + j;
    scK0[j] = scale[k0]; shK0[j] = shift[k0]; baK0[j] = ba[k0];
    scK1[j] = scale[k1]; shK1[j] = shift[k1]; baK1[j] = ba[k1];
  }

  float accS0 = 0.f, accQ0 = 0.f, accS1 = 0.f, accQ1 = 0.f;

  const int wid = blockIdx.x * (blockDim.x >> 5) + wslot;
  const int nw  = gridDim.x * (blockDim.x >> 5);
  __bf16* myTile = &lds_r1[wslot][0];

  for (int tile = wid; tile < NT; tile += nw) {
    const size_t rowbase = (size_t)tile * 16;
    const float* hp = hin + (rowbase + l15) * DD;
    const float* ap = agg + (rowbase + l15) * DD;
    v16bf a2;   // A-fragment of r1 for the second GEMM
    if constexpr (DO_GEMM1) {
      // ---- GEMM1: t = bn(hin)+agg, single K=32 WMMA step per col tile ----
      v16bf a;
#pragma unroll
      for (int j = 0; j < 8; ++j) {
        const int k0 = 8 * hh + j;
        const int k1 = 16 + 8 * hh + j;
        a[j]     = (__bf16)(fmaf(hp[k0], scK0[j], shK0[j]) + ap[k0]);
        a[8 + j] = (__bf16)(fmaf(hp[k1], scK1[j], shK1[j]) + ap[k1]);
      }
      v8f c0, c1;
#pragma unroll
      for (int i = 0; i < 8; ++i) { c0[i] = ba_c0; c1[i] = ba_c1; }
      c0 = WMMA_BF16(a, waF[0], c0);
      c1 = WMMA_BF16(a, waF[1], c1);
      // relu -> per-wave LDS tile (accumulator layout -> row-major)
#pragma unroll
      for (int i = 0; i < 8; ++i) {
        myTile[(i + 8 * hh) * LDSS + l15]      = (__bf16)fmaxf(c0[i], 0.0f);
        myTile[(i + 8 * hh) * LDSS + 16 + l15] = (__bf16)fmaxf(c1[i], 0.0f);
      }
      asm volatile("s_wait_dscnt 0x0" ::: "memory");  // cross-lane LDS RAW
      const __bf16* r = myTile + l15 * LDSS;
#pragma unroll
      for (int j = 0; j < 8; ++j) {
        a2[j]     = r[8 * hh + j];
        a2[8 + j] = r[16 + 8 * hh + j];
      }
    } else {
      // layer-1 path: r1 = relu(u + agg_u + b1a), built directly at frag positions
#pragma unroll
      for (int j = 0; j < 8; ++j) {
        const int k0 = 8 * hh + j;
        const int k1 = 16 + 8 * hh + j;
        a2[j]     = (__bf16)fmaxf(fmaf(hp[k0], scK0[j], shK0[j]) + ap[k0] + baK0[j], 0.0f);
        a2[8 + j] = (__bf16)fmaxf(fmaf(hp[k1], scK1[j], shK1[j]) + ap[k1] + baK1[j], 0.0f);
      }
    }
    // ---- GEMM2: out = relu(r1@Wb + bb), fused stats ----
    v8f d0, d1;
#pragma unroll
    for (int i = 0; i < 8; ++i) { d0[i] = bb_c0; d1[i] = bb_c1; }
    d0 = WMMA_BF16(a2, wbF[0], d0);
    d1 = WMMA_BF16(a2, wbF[1], d1);
    float* op = out + rowbase * DD;
#pragma unroll
    for (int i = 0; i < 8; ++i) {
      float v0 = fmaxf(d0[i], 0.0f);
      float v1 = fmaxf(d1[i], 0.0f);
      op[(i + 8 * hh) * DD + l15]      = v0;
      op[(i + 8 * hh) * DD + 16 + l15] = v1;
      accS0 += v0; accQ0 += v0 * v0;
      accS1 += v1; accQ1 += v1 * v1;
    }
  }

  // Block-level BN stat reduction, then 64 global atomics per block.
  if (threadIdx.x < DD) { s_sum[threadIdx.x] = 0.f; s_sq[threadIdx.x] = 0.f; }
  __syncthreads();
  atomicAdd(&s_sum[l15], accS0);      atomicAdd(&s_sum[16 + l15], accS1);
  atomicAdd(&s_sq[l15],  accQ0);      atomicAdd(&s_sq[16 + l15],  accQ1);
  __syncthreads();
  if (threadIdx.x < DD) {
    atomicAdd(&stat_sum[threadIdx.x], s_sum[threadIdx.x]);
    atomicAdd(&stat_sq[threadIdx.x],  s_sq[threadIdx.x]);
  }
}

// ---------------------------------------------------------------------
// BatchNorm finalize: produces scale/shift consumed by the NEXT kernels
// that read this layer's output (scatter / mlp hin / pool).
// ---------------------------------------------------------------------
__global__ void k_bn_finalize(const float* __restrict__ sum, const float* __restrict__ sq,
                              const float* __restrict__ gamma, const float* __restrict__ beta,
                              float* __restrict__ scale, float* __restrict__ shift) {
  int c = threadIdx.x;
  if (c >= DD) return;
  const float invN = 1.0f / (float)NN;
  float m = sum[c] * invN;
  float v = sq[c] * invN - m * m;
  float sc = gamma[c] * rsqrtf(v + BN_EPS);
  scale[c] = sc;
  shift[c] = beta[c] - m * sc;
}

// ---------------------------------------------------------------------
// global_add_pool with fused final BN: pooled[batch[n]] += bn(h[n])
// ---------------------------------------------------------------------
__global__ void k_pool(const float* __restrict__ h, const int* __restrict__ batch,
                       float* __restrict__ pooled,
                       const float* __restrict__ scale, const float* __restrict__ shift) {
  const int lane = threadIdx.x & 31;
  const float sc = scale[lane];
  const float sh = shift[lane];
  const int w  = (blockIdx.x * blockDim.x + threadIdx.x) >> 5;
  const int nw = (gridDim.x * blockDim.x) >> 5;
  for (int node = w; node < NN; node += nw) {
    int g = batch[node];
    const float v = fmaf(h[(size_t)node * DD + lane], sc, sh);
    atomicAdd(&pooled[(size_t)g * DD + lane], v);
  }
}

// ---------------------------------------------------------------------
// FC head (tiny: 1000 graphs)
// ---------------------------------------------------------------------
__global__ void k_fc1(const float* __restrict__ pooled, const float* __restrict__ w,
                      const float* __restrict__ b, float* __restrict__ outp) {
  int i = blockIdx.x * blockDim.x + threadIdx.x;
  if (i >= GG * DD) return;
  int g = i >> 5, j = i & (DD - 1);
  float s = b[j];
#pragma unroll
  for (int k = 0; k < DD; ++k) s += pooled[g * DD + k] * w[k * DD + j];
  outp[i] = fmaxf(s, 0.0f);
}

__global__ void k_fc2_logsoftmax(const float* __restrict__ f1, const float* __restrict__ w,
                                 const float* __restrict__ b, float* __restrict__ outp) {
  int g = blockIdx.x * blockDim.x + threadIdx.x;
  if (g >= GG) return;
  float l[CC];
#pragma unroll
  for (int c = 0; c < CC; ++c) {
    float s = b[c];
#pragma unroll
    for (int k = 0; k < DD; ++k) s += f1[g * DD + k] * w[k * CC + c];
    l[c] = s;
  }
  float m = l[0];
#pragma unroll
  for (int c = 1; c < CC; ++c) m = fmaxf(m, l[c]);
  float e = 0.f;
#pragma unroll
  for (int c = 0; c < CC; ++c) e += expf(l[c] - m);
  float lse = logf(e);
#pragma unroll
  for (int c = 0; c < CC; ++c) outp[g * CC + c] = l[c] - m - lse;
}

// ---------------------------------------------------------------------
static inline size_t al256(size_t x) { return (x + 255) & ~(size_t)255; }

extern "C" void kernel_launch(void* const* d_in, const int* in_sizes, int n_in,
                              void* d_out, int out_size, void* d_ws, size_t ws_size,
                              hipStream_t stream) {
  (void)in_sizes; (void)n_in; (void)out_size; (void)ws_size;
  const float* x     = (const float*)d_in[0];
  const int*   edge  = (const int*)d_in[1];
  const int*   srcI  = edge;
  const int*   dstI  = edge + EE;
  const int*   batch = (const int*)d_in[2];
  const float* W1a   = (const float*)d_in[3];
  const float* b1a   = (const float*)d_in[4];
  const float* W1b   = (const float*)d_in[5];
  const float* b1b   = (const float*)d_in[6];
  const float* Wa    = (const float*)d_in[7];
  const float* ba    = (const float*)d_in[8];
  const float* Wb    = (const float*)d_in[9];
  const float* bb    = (const float*)d_in[10];
  const float* gamma = (const float*)d_in[11];
  const float* beta  = (const float*)d_in[12];
  const float* fc1w  = (const float*)d_in[13];
  const float* fc1b  = (const float*)d_in[14];
  const float* fc2w  = (const float*)d_in[15];
  const float* fc2b  = (const float*)d_in[16];

  // --- workspace layout ---
  char* p = (char*)d_ws;
  float* bufA = (float*)p;  p += al256((size_t)NN * DD * sizeof(float));
  float* bufB = (float*)p;  p += al256((size_t)NN * DD * sizeof(float));
  float* agg  = (float*)p;  p += al256((size_t)NN * DD * sizeof(float));
  float* stat = (float*)p;  p += al256(64 * sizeof(float));   // sum[32] | sumsq[32]
  float* scl  = (float*)p;  p += al256(DD * sizeof(float));
  float* shf  = (float*)p;  p += al256(DD * sizeof(float));
  float* pooled = (float*)p; p += al256((size_t)GG * DD * sizeof(float));
  float* f1out  = (float*)p; p += al256((size_t)GG * DD * sizeof(float));
  __bf16* w1a_t = (__bf16*)p; p += al256((size_t)DD * FF * sizeof(__bf16));
  __bf16* w1b_t = (__bf16*)p; p += al256((size_t)DD * DD * sizeof(__bf16));
  __bf16* wa_t  = (__bf16*)p; p += al256((size_t)4 * DD * DD * sizeof(__bf16));
  __bf16* wb_t  = (__bf16*)p; p += al256((size_t)4 * DD * DD * sizeof(__bf16));

  // --- weight prep (bf16, transposed to [Nout][K]) ---
  k_transpose_bf16<<<(DD * FF + 255) / 256, 256, 0, stream>>>(W1a, w1a_t, 1, FF, DD);
  k_transpose_bf16<<<(DD * DD + 255) / 256, 256, 0, stream>>>(W1b, w1b_t, 1, DD, DD);
  k_transpose_bf16<<<(4 * DD * DD + 255) / 256, 256, 0, stream>>>(Wa, wa_t, 4, DD, DD);
  k_transpose_bf16<<<(4 * DD * DD + 255) / 256, 256, 0, stream>>>(Wb, wb_t, 4, DD, DD);

  // --- layer-1 linearity trick: u = x @ W1a (WMMA, K=128) ---
  k_gemm_xw<<<784, 128, 0, stream>>>(x, w1a_t, bufA);

  // BN affine starts as identity (u feeds layer 1 directly).
  k_init_affine<<<1, 32, 0, stream>>>(scl, shf);

  const int zeroGrid = (NN * DD + 255) / 256;
  float* cur = bufA;
  float* nxt = bufB;
  for (int layer = 0; layer < 5; ++layer) {
    k_zero<<<zeroGrid, 256, 0, stream>>>(agg, NN * DD);
    k_zero<<<1, 64, 0, stream>>>(stat, 64);
    // agg = sum over edges of bn(cur)[src]  (affine fused into gather)
    k_scatter<<<2048, 256, 0, stream>>>(srcI, dstI, cur, agg, scl, shf);
    if (layer == 0) {
      k_gin_mlp<false><<<784, 128, 0, stream>>>(cur, agg, scl, shf, w1b_t /*unused*/, b1a,
                                                w1b_t, b1b, nxt, stat, stat + 32);
    } else {
      const int i = layer - 1;
      k_gin_mlp<true><<<784, 128, 0, stream>>>(cur, agg, scl, shf,
                                               wa_t + (size_t)i * DD * DD, ba + i * DD,
                                               wb_t + (size_t)i * DD * DD, bb + i * DD,
                                               nxt, stat, stat + 32);
    }
    // Produce this layer's BN affine for the NEXT consumers of `nxt`.
    k_bn_finalize<<<1, 32, 0, stream>>>(stat, stat + 32, gamma + layer * DD,
                                        beta + layer * DD, scl, shf);
    float* t = cur; cur = nxt; nxt = t;
  }

  // --- pooling (final BN fused) + FC head ---
  k_zero<<<(GG * DD + 255) / 256, 256, 0, stream>>>(pooled, GG * DD);
  k_pool<<<1024, 256, 0, stream>>>(cur, batch, pooled, scl, shf);
  k_fc1<<<(GG * DD + 255) / 256, 256, 0, stream>>>(pooled, fc1w, fc1b, f1out);
  k_fc2_logsoftmax<<<(GG + 255) / 256, 256, 0, stream>>>(f1out, fc2w, fc2b, (float*)d_out);
}
